// WeatherLSTM_68719477197
// MI455X (gfx1250) — 8-bit matrix operand order repaired and checked by reading, compiled, not run on hardware
//
#include <hip/hip_runtime.h>

#define DEVFN __device__ __forceinline__

typedef __attribute__((ext_vector_type(8)))  int   v8i;
typedef __attribute__((ext_vector_type(16))) int   v16i;
typedef __attribute__((ext_vector_type(8)))  float v8f;

namespace {
constexpr int kIN = 8, kH = 256, kOUT = 24, kB = 2048, kT = 256;
constexpr int k4H = 4 * kH;
constexpr int kSteps = kT + (kOUT - 1);
constexpr int kRowsPerBlock = 16;
constexpr int kThreads = 512;

constexpr int WH_COL_STRIDE = 272;
constexpr int WH_GATE  = kH * WH_COL_STRIDE;
constexpr int WH_BYTES = 3 * WH_GATE;
constexpr int STG_ROW   = 328;
constexpr int STG_BUF   = kRowsPerBlock * STG_ROW;
constexpr int STG_BYTES = 2 * STG_BUF;
constexpr int WY_OFF    = WH_BYTES + STG_BYTES;
constexpr int LDS_BYTES = WY_OFF + kH * 4;
}

DEVFN unsigned fp8_enc1(float x) {
  union { float f; unsigned u; } v; v.f = x;
  unsigned s = (v.u >> 24) & 0x80u;
  unsigned exf = (v.u >> 23) & 0xFFu;
  if (exf == 0xFFu) return s | 0x7Eu;
  int e = (int)exf - 127;
  unsigned m = v.u & 0x7FFFFFu;
  if (e >= -6) {
    if (e > 8) return s | 0x7Eu;
    unsigned bits = (((unsigned)(e + 7)) << 3) | (m >> 20);
    bits += (m >> 19) & 1u;
    if (bits > 0x7Eu) bits = 0x7Eu;
    return s | bits;
  }
  float mag = __builtin_fabsf(x) * 512.0f;
  unsigned q = (unsigned)(mag + 0.5f);
  if (q > 7u) q = 8u;
  return s | q;
}

DEVFN int pk2_fp8(float a, float b) {
#if __has_builtin(__builtin_amdgcn_cvt_pk_fp8_f32)
  return __builtin_amdgcn_cvt_pk_fp8_f32(a, b, 0, false);
#else
  return (int)(fp8_enc1(a) | (fp8_enc1(b) << 8));
#endif
}

DEVFN float fp8_val(int byte0) {
#if __has_builtin(__builtin_amdgcn_cvt_f32_fp8)
  return __builtin_amdgcn_cvt_f32_fp8(byte0, 0);
#else
  unsigned bb = (unsigned)byte0 & 0xFFu;
  unsigned E = (bb >> 3) & 0xFu, M = bb & 7u;
  float v = E ? __builtin_ldexpf((float)(8u + M), (int)E - 10)
              : (float)M * (1.0f / 512.0f);
  return (bb & 0x80u) ? -v : v;
#endif
}

DEVFN float fast_sigmoid(float x) {
  return __builtin_amdgcn_rcpf(1.0f + __builtin_amdgcn_exp2f(-1.44269504f * x));
}
DEVFN float fast_tanh(float x) {
  return 1.0f - 2.0f * __builtin_amdgcn_rcpf(1.0f + __builtin_amdgcn_exp2f(2.88539008f * x));
}

DEVFN v16i load_a128(const char* p) {
  v16i r;
#pragma unroll
  for (int i = 0; i < 8; ++i) {
    int2 q = *(const int2*)(p + i * 16);
    r[2 * i] = q.x; r[2 * i + 1] = q.y;
  }
  return r;
}
DEVFN v8i load_a64(const char* p) {
  v8i r;
#pragma unroll
  for (int i = 0; i < 4; ++i) {
    int2 q = *(const int2*)(p + i * 16);
    r[2 * i] = q.x; r[2 * i + 1] = q.y;
  }
  return r;
}
DEVFN v16i load_b128(const char* p) {
  v16i r;
#pragma unroll
  for (int i = 0; i < 8; ++i) {
    int2 q = *(const int2*)(p + i * 16);
    r[2 * i] = q.x; r[2 * i + 1] = q.y;
  }
  return r;
}

__global__ __launch_bounds__(kThreads, 1)
void lstm_step_fp8_wmma(const float* __restrict__ x0,
                        const float* __restrict__ x1,
                        const float* __restrict__ Wx,
                        const float* __restrict__ Wh,
                        const float* __restrict__ bg,
                        const float* __restrict__ Wy,
                        const float* __restrict__ by,
                        const float* __restrict__ dw,
                        const float* __restrict__ db,
                        float* __restrict__ out)
{
  extern __shared__ char smem[];
  char*  whl = smem;
  char*  stg = smem + WH_BYTES;
  float* wyl = (float*)(smem + WY_OFF);

  const int tid = threadIdx.x;
  const int wv  = tid >> 5;
  const int ln  = tid & 31;
  const int lm  = ln & 15;
  const int hi  = ln >> 4;
  const int mbase = blockIdx.x * kRowsPerBlock;

  const int goff[3] = {0, 2 * kH, 3 * kH};

  for (int idx = tid; idx < 3 * kH * (kH / 2); idx += kThreads) {
    int n  = idx & 255;
    int kp = (idx >> 8) & 127;
    int g  = idx >> 15;
    int k  = kp * 2;
    int col = goff[g] + n;
    float w0 = Wh[(size_t)k * k4H + col];
    float w1 = Wh[(size_t)(k + 1) * k4H + col];
    int p = pk2_fp8(w0, w1);
    *(short*)(whl + g * WH_GATE + n * WH_COL_STRIDE + k) = (short)(p & 0xFFFF);
  }
  for (int idx = tid * 4; idx < STG_BYTES; idx += kThreads * 4) *(int*)(stg + idx) = 0;
  for (int k = tid; k < kH; k += kThreads) wyl[k] = Wy[k];
  if (tid < kRowsPerBlock * kIN) {
    int m = tid >> 3, c = tid & 7;
    float xv = x0[((size_t)(mbase + m)) * kT * kIN + c];
    stg[m * STG_ROW + kH + c] = (char)(pk2_fp8(xv, 0.0f) & 0xFF);
  }
  __syncthreads();

  const int ncol = wv * 16 + lm;
  v16i Bres[3][2];
  v8i  Wxf[3];
  float bias[3];
#pragma unroll
  for (int g = 0; g < 3; ++g) {
    const char* cb = whl + g * WH_GATE + ncol * WH_COL_STRIDE + hi * 8;
    Bres[g][0] = load_b128(cb);
    Bres[g][1] = load_b128(cb + 128);
    v8i f;
#pragma unroll
    for (int i = 0; i < 8; ++i) f[i] = 0;
    if (hi == 0) {
      int col = goff[g] + ncol;
      int p01 = pk2_fp8(Wx[0 * k4H + col], Wx[1 * k4H + col]) & 0xFFFF;
      int p23 = pk2_fp8(Wx[2 * k4H + col], Wx[3 * k4H + col]) & 0xFFFF;
      int p45 = pk2_fp8(Wx[4 * k4H + col], Wx[5 * k4H + col]) & 0xFFFF;
      int p67 = pk2_fp8(Wx[6 * k4H + col], Wx[7 * k4H + col]) & 0xFFFF;
      f[0] = p01 | (p23 << 16);
      f[1] = p45 | (p67 << 16);
    }
    Wxf[g]  = f;
    bias[g] = bg[goff[g] + ncol];
  }
  __syncthreads();

  int cur = 0;
  for (int t = 0; t < kSteps; ++t) {
    const char* ab = stg + cur * STG_BUF + lm * STG_ROW + hi * 8;
    v16i A0 = load_a128(ab);
    v16i A1 = load_a128(ab + 128);
    v8i  Ax = load_a64 (ab + 256);

    v8f acc[3];
#pragma unroll
    for (int g = 0; g < 3; ++g) {
      float bv = bias[g];
      v8f c = {bv, bv, bv, bv, bv, bv, bv, bv};
      c = __builtin_amdgcn_wmma_f32_16x16x64_fp8_fp8 (Ax, Wxf[g],    (short)0, c, false, false);
      c = __builtin_amdgcn_wmma_f32_16x16x128_fp8_fp8(A0, Bres[g][0], (short)0, c, false, false);
      c = __builtin_amdgcn_wmma_f32_16x16x128_fp8_fp8(A1, Bres[g][1], (short)0, c, false, false);
      acc[g] = c;
    }

    char* nb = stg + (cur ^ 1) * STG_BUF;
#pragma unroll
    for (int r = 0; r < 8; r += 2) {
      float h0 = fast_sigmoid(acc[1][r])     * fast_tanh(fast_sigmoid(acc[0][r])     * fast_tanh(acc[2][r]));
      float h1 = fast_sigmoid(acc[1][r + 1]) * fast_tanh(fast_sigmoid(acc[0][r + 1]) * fast_tanh(acc[2][r + 1]));
      int p = pk2_fp8(h0, h1);
      int m0 = hi * 8 + r;
      nb[m0 * STG_ROW + ncol]       = (char)(p & 0xFF);
      nb[(m0 + 1) * STG_ROW + ncol] = (char)((p >> 8) & 0xFF);
    }

    if (t < kSteps - 1 && tid < kRowsPerBlock * kIN) {
      int m = tid >> 3, c = tid & 7, tn = t + 1;
      float xv;
      if (tn < kT) {
        xv = x0[((size_t)(mbase + m)) * kT * kIN + (size_t)tn * kIN + c];
      } else {
        int s = tn - kT;
        xv = x1[(size_t)(mbase + m) * (kOUT - 1) + s] * dw[s * kIN + c] + db[s * kIN + c];
      }
      nb[m * STG_ROW + kH + c] = (char)(pk2_fp8(xv, 0.0f) & 0xFF);
    }

    __syncthreads();

    if (t >= kT - 1 && tid < kRowsPerBlock * kIN) {
      int m = tid >> 3, ch = tid & 7;
      float s = 0.0f;
#pragma unroll 8
      for (int k = 0; k < 32; ++k) {
        int kk = ch * 32 + k;
        s += fp8_val((unsigned char)nb[m * STG_ROW + kk]) * wyl[kk];
      }
      s += __shfl_down(s, 4, 8);
      s += __shfl_down(s, 2, 8);
      s += __shfl_down(s, 1, 8);
      if (ch == 0) out[(size_t)(mbase + m) * kOUT + (t - (kT - 1))] = s + by[0];
    }

    cur ^= 1;
  }
}

extern "C" void kernel_launch(void* const* d_in, const int* in_sizes, int n_in,
                              void* d_out, int out_size, void* d_ws, size_t ws_size,
                              hipStream_t stream) {
  (void)in_sizes; (void)n_in; (void)out_size; (void)d_ws; (void)ws_size;
  const float* x0 = (const float*)d_in[0];
  const float* x1 = (const float*)d_in[1];
  const float* Wx = (const float*)d_in[2];
  const float* Wh = (const float*)d_in[3];
  const float* bg = (const float*)d_in[4];
  const float* Wy = (const float*)d_in[5];
  const float* by = (const float*)d_in[6];
  const float* dw = (const float*)d_in[7];
  const float* db = (const float*)d_in[8];
  float* out = (float*)d_out;

  (void)hipFuncSetAttribute((const void*)lstm_step_fp8_wmma,
                            hipFuncAttributeMaxDynamicSharedMemorySize, LDS_BYTES);

  lstm_step_fp8_wmma<<<dim3(kB / kRowsPerBlock), dim3(kThreads), LDS_BYTES, stream>>>(
      x0, x1, Wx, Wh, bg, Wy, by, dw, db, out);
}
